// LGNNLayer_19069654794985
// MI455X (gfx1250) — matrix operand order repaired and checked by reading, not compiled, not run on hardware
//
#include <hip/hip_runtime.h>
#include <math.h>

// ---------------------------------------------------------------------------
// Types for CDNA5 WMMA (wave32): v_wmma_f32_16x16x32_bf16
// ---------------------------------------------------------------------------
typedef __attribute__((ext_vector_type(16))) __bf16 bf16x16;
typedef __attribute__((ext_vector_type(8)))  __bf16 bf16x8;
typedef __attribute__((ext_vector_type(8)))  float  f32x8;

__device__ __forceinline__ float gelu_exact(float v) {
  return 0.5f * v * (1.0f + erff(v * 0.70710678118654752440f));
}

// ---------------------------------------------------------------------------
// Elementwise helpers
// ---------------------------------------------------------------------------
__global__ void k_zero_f32(float* __restrict__ p, long n) {
  long i = (long)blockIdx.x * blockDim.x + threadIdx.x;
  if (i < n) p[i] = 0.0f;
}

__global__ void k_cvt_bf16(const float* __restrict__ in, __bf16* __restrict__ out, long n) {
  long i = (long)blockIdx.x * blockDim.x + threadIdx.x;
  if (i < n) out[i] = (__bf16)in[i];
}

// WT[n][k] = (bf16) W[k][n]   (weights are tiny; done once per core)
__global__ void k_transpose_cvt(const float* __restrict__ W, __bf16* __restrict__ WT,
                                int K, int N) {
  int i = blockIdx.x * blockDim.x + threadIdx.x;
  if (i >= K * N) return;
  int k = i / N, n = i - k * N;
  WT[(size_t)n * K + k] = (__bf16)W[i];
}

// ---------------------------------------------------------------------------
// Graph scatter / gather kernels (D = 128, 32 threads x float4 per edge).
// unsafeAtomicAdd -> native global_atomic_add_f32 (no CAS loop).
// ---------------------------------------------------------------------------
__global__ void k_scatter_add(const float* __restrict__ zin, const int* __restrict__ src,
                              const int* __restrict__ dst, float* __restrict__ zout, int nE) {
  long t = (long)blockIdx.x * blockDim.x + threadIdx.x;
  long total = (long)nE * 32;
  if (t >= total) return;
  int e = (int)(t >> 5);
  int c = ((int)t & 31) * 4;
  float4 v = *(const float4*)(zin + (size_t)src[e] * 128 + c);
  float* o = zout + (size_t)dst[e] * 128 + c;
  unsafeAtomicAdd(o + 0, v.x);
  unsafeAtomicAdd(o + 1, v.y);
  unsafeAtomicAdd(o + 2, v.z);
  unsafeAtomicAdd(o + 3, v.w);
}

// node_inc = segment_sum(lg_x, src) + segment_sum(lg_x, dst)
__global__ void k_inc_scatter(const float* __restrict__ lgx, const int* __restrict__ src,
                              const int* __restrict__ dst, float* __restrict__ inc, int nE) {
  long t = (long)blockIdx.x * blockDim.x + threadIdx.x;
  long total = (long)nE * 32;
  if (t >= total) return;
  int e = (int)(t >> 5);
  int c = ((int)t & 31) * 4;
  float4 v = *(const float4*)(lgx + (size_t)e * 128 + c);
  float* o0 = inc + (size_t)src[e] * 128 + c;
  float* o1 = inc + (size_t)dst[e] * 128 + c;
  unsafeAtomicAdd(o0 + 0, v.x); unsafeAtomicAdd(o0 + 1, v.y);
  unsafeAtomicAdd(o0 + 2, v.z); unsafeAtomicAdd(o0 + 3, v.w);
  unsafeAtomicAdd(o1 + 0, v.x); unsafeAtomicAdd(o1 + 1, v.y);
  unsafeAtomicAdd(o1 + 2, v.z); unsafeAtomicAdd(o1 + 3, v.w);
}

// edge_inc[e] = x[src[e]] + x[dst[e]]
__global__ void k_edge_inc(const float* __restrict__ x, const int* __restrict__ src,
                           const int* __restrict__ dst, float* __restrict__ out, int nE) {
  long t = (long)blockIdx.x * blockDim.x + threadIdx.x;
  long total = (long)nE * 32;
  if (t >= total) return;
  int e = (int)(t >> 5);
  int c = ((int)t & 31) * 4;
  float4 a = *(const float4*)(x + (size_t)src[e] * 128 + c);
  float4 b = *(const float4*)(x + (size_t)dst[e] * 128 + c);
  float4 r = make_float4(a.x + b.x, a.y + b.y, a.z + b.z, a.w + b.w);
  *(float4*)(out + (size_t)e * 128 + c) = r;
}

// ---------------------------------------------------------------------------
// Fragment loaders (CDNA5 ISA 7.12.2 layouts).
//   A (16x32 bf16): lane l (m=l&15, h=l>>4) holds k = k0 + {h*8..+7, 16+h*8..+7}
//   B (32x16 bf16): lane l (n=l&15, h=l>>4) holds k = k0 + {h*8..+7, 16+h*8..+7}
// ---------------------------------------------------------------------------
__device__ __forceinline__ bf16x16 load_a_frag(const __bf16* __restrict__ row, int k0, int h) {
  bf16x8 lo = *(const bf16x8*)(row + k0 + h * 8);
  bf16x8 hi = *(const bf16x8*)(row + k0 + 16 + h * 8);
  bf16x16 r;
#pragma unroll
  for (int j = 0; j < 8; ++j) { r[j] = lo[j]; r[j + 8] = hi[j]; }
  return r;
}

__device__ __forceinline__ bf16x16 load_b_frag(const __bf16* __restrict__ row, int k0, int h) {
  bf16x8 lo = *(const bf16x8*)(row + k0 + h * 8);
  bf16x8 hi = *(const bf16x8*)(row + k0 + 16 + h * 8);
  bf16x16 r;
#pragma unroll
  for (int j = 0; j < 8; ++j) { r[j] = lo[j]; r[j + 8] = hi[j]; }
  return r;
}

// ---------------------------------------------------------------------------
// WMMA GEMM:  C[M,N] (+)= A[M,K](bf16,row-major) * B[K,N]  with BT[N,K] bf16.
// Register-blocked: one wave owns a 32x64 output block (2x4 subtiles of
// 16x16), so each 32-wide K-chunk does 6 fragment loads (12 x b128) feeding
// 8 v_wmma_f32_16x16x32_bf16 ops (1.5 loads/WMMA, A reused 4x, B reused 2x).
// C/D layout: lane l, vgpr v -> row v+8*(l>>4), col l&15.
// Epilogue: optional bias, accumulate into Cf, exact GELU, bf16 store (Cbf).
// ---------------------------------------------------------------------------
__global__ __launch_bounds__(256) void k_gemm_bf16_wmma(
    const __bf16* __restrict__ A, const __bf16* __restrict__ BT,
    const float* __restrict__ bias,
    float* __restrict__ Cf, __bf16* __restrict__ Cbf,
    int M, int N, int K, int accumulate, int do_gelu) {
  const int wv = blockIdx.x * (blockDim.x >> 5) + (threadIdx.x >> 5);
  const int tnc = N >> 6;                    // 64-wide column groups
  const int total = (M >> 5) * tnc;          // 32-row blocks
  if (wv >= total) return;                   // whole-wave uniform: EXEC stays all-ones
  const int tm = wv / tnc;
  const int tn = wv - tm * tnc;
  const int lane = threadIdx.x & 31;
  const int g = lane & 15;
  const int h = lane >> 4;

  const __bf16* arow0 = A + (size_t)(tm * 32 + g) * K;         // m-subtile 0
  const __bf16* arow1 = arow0 + (size_t)16 * K;                // m-subtile 1
  const __bf16* brow0 = BT + (size_t)(tn * 64 + g) * K;        // n-subtile 0
  const size_t bstep = (size_t)16 * K;

  f32x8 acc[2][4] = {};
  for (int k0 = 0; k0 < K; k0 += 32) {
    __builtin_prefetch((const void*)(arow0 + k0 + 32), 0, 0);  // global_prefetch_b8
    __builtin_prefetch((const void*)(arow1 + k0 + 32), 0, 0);
    bf16x16 a0 = load_a_frag(arow0, k0, h);
    bf16x16 a1 = load_a_frag(arow1, k0, h);
    bf16x16 b0 = load_b_frag(brow0, k0, h);
    bf16x16 b1 = load_b_frag(brow0 + bstep, k0, h);
    bf16x16 b2 = load_b_frag(brow0 + 2 * bstep, k0, h);
    bf16x16 b3 = load_b_frag(brow0 + 3 * bstep, k0, h);
    acc[0][0] = __builtin_amdgcn_wmma_f32_16x16x32_bf16(false, a0, false, b0, (short)0, acc[0][0], false, false);
    acc[0][1] = __builtin_amdgcn_wmma_f32_16x16x32_bf16(false, a0, false, b1, (short)0, acc[0][1], false, false);
    acc[0][2] = __builtin_amdgcn_wmma_f32_16x16x32_bf16(false, a0, false, b2, (short)0, acc[0][2], false, false);
    acc[0][3] = __builtin_amdgcn_wmma_f32_16x16x32_bf16(false, a0, false, b3, (short)0, acc[0][3], false, false);
    acc[1][0] = __builtin_amdgcn_wmma_f32_16x16x32_bf16(false, a1, false, b0, (short)0, acc[1][0], false, false);
    acc[1][1] = __builtin_amdgcn_wmma_f32_16x16x32_bf16(false, a1, false, b1, (short)0, acc[1][1], false, false);
    acc[1][2] = __builtin_amdgcn_wmma_f32_16x16x32_bf16(false, a1, false, b2, (short)0, acc[1][2], false, false);
    acc[1][3] = __builtin_amdgcn_wmma_f32_16x16x32_bf16(false, a1, false, b3, (short)0, acc[1][3], false, false);
  }

#pragma unroll
  for (int nj = 0; nj < 4; ++nj) {
    const int col = tn * 64 + nj * 16 + g;
    const float bv = bias ? bias[col] : 0.0f;
#pragma unroll
    for (int mi = 0; mi < 2; ++mi) {
#pragma unroll
      for (int v = 0; v < 8; ++v) {
        const int row = tm * 32 + mi * 16 + h * 8 + v;
        const size_t idx = (size_t)row * N + col;
        float val = acc[mi][nj][v] + bv;
        if (accumulate) val += Cf[idx];
        if (do_gelu) val = gelu_exact(val);
        if (Cf)  Cf[idx]  = val;
        if (Cbf) Cbf[idx] = (__bf16)val;
      }
    }
  }
}

// ---------------------------------------------------------------------------
// Fused  h = LayerNorm(x + gelu(pre))  -> fp32 h and bf16 h (next GEMM input)
// One row (D=128) per 128-thread block.
// ---------------------------------------------------------------------------
__global__ __launch_bounds__(128) void k_gelu_res_ln(
    const float* __restrict__ x, const float* __restrict__ pre,
    const float* __restrict__ gam, const float* __restrict__ bet,
    float* __restrict__ hf, __bf16* __restrict__ hb) {
  const int d = threadIdx.x;
  const size_t i = (size_t)blockIdx.x * 128 + d;
  float t = x[i] + gelu_exact(pre[i]);
  __shared__ float s[128];
  s[d] = t; __syncthreads();
  for (int o = 64; o > 0; o >>= 1) { if (d < o) s[d] += s[d + o]; __syncthreads(); }
  float mu = s[0] * (1.0f / 128.0f);
  __syncthreads();
  float df = t - mu;
  s[d] = df * df; __syncthreads();
  for (int o = 64; o > 0; o >>= 1) { if (d < o) s[d] += s[d + o]; __syncthreads(); }
  float y = df * rsqrtf(s[0] * (1.0f / 128.0f) + 1e-5f) * gam[d] + bet[d];
  hf[i] = y;
  hb[i] = (__bf16)y;
}

// out = LayerNorm(a + b)
__global__ __launch_bounds__(128) void k_res_ln(
    const float* __restrict__ a, const float* __restrict__ b,
    const float* __restrict__ gam, const float* __restrict__ bet,
    float* __restrict__ out) {
  const int d = threadIdx.x;
  const size_t i = (size_t)blockIdx.x * 128 + d;
  float t = a[i] + b[i];
  __shared__ float s[128];
  s[d] = t; __syncthreads();
  for (int o = 64; o > 0; o >>= 1) { if (d < o) s[d] += s[d + o]; __syncthreads(); }
  float mu = s[0] * (1.0f / 128.0f);
  __syncthreads();
  float df = t - mu;
  s[d] = df * df; __syncthreads();
  for (int o = 64; o > 0; o >>= 1) { if (d < o) s[d] += s[d + o]; __syncthreads(); }
  out[i] = df * rsqrtf(s[0] * (1.0f / 128.0f) + 1e-5f) * gam[d] + bet[d];
}

// ---------------------------------------------------------------------------
// Host-side orchestration
// ---------------------------------------------------------------------------
struct CoreParams {
  const float *W_ff1, *W_ff2, *W_fuse, *W_khops, *W_self;
  const float *b_ff1, *b_ff2, *b_fuse, *b_khops, *b_self;
  const float *ln1_b, *ln1_g, *ln2_b, *ln2_g;
};

static CoreParams unpack_params(void* const* d_in, int base) {
  // JAX pytree (sorted dict key) order
  CoreParams p;
  p.W_ff1   = (const float*)d_in[base + 0];
  p.W_ff2   = (const float*)d_in[base + 1];
  p.W_fuse  = (const float*)d_in[base + 2];
  p.W_khops = (const float*)d_in[base + 3];
  p.W_self  = (const float*)d_in[base + 4];
  p.b_ff1   = (const float*)d_in[base + 5];
  p.b_ff2   = (const float*)d_in[base + 6];
  p.b_fuse  = (const float*)d_in[base + 7];
  p.b_khops = (const float*)d_in[base + 8];
  p.b_self  = (const float*)d_in[base + 9];
  p.ln1_b   = (const float*)d_in[base + 10];
  p.ln1_g   = (const float*)d_in[base + 11];
  p.ln2_b   = (const float*)d_in[base + 12];
  p.ln2_g   = (const float*)d_in[base + 13];
  return p;
}

static inline void launch_gemm(hipStream_t s, const __bf16* A, const __bf16* BT,
                               const float* bias, float* Cf, __bf16* Cbf,
                               int M, int N, int K, int acc, int gl) {
  int tiles = (M / 32) * (N / 64);   // one wave per 32x64 block
  int blocks = (tiles + 7) / 8;      // 8 waves (256 threads) per block
  k_gemm_bf16_wmma<<<blocks, 256, 0, s>>>(A, BT, bias, Cf, Cbf, M, N, K, acc, gl);
}

static void run_core(hipStream_t st, const CoreParams& p,
                     const float* x, const int* esrc, const int* edst, int nE, int n,
                     const float* inc,
                     __bf16* bfA, __bf16* bfH, float* z0, float* z1,
                     float* pre, float* h1, __bf16* wT, float* out) {
  // ---- pack weights (fp32 [K,N] -> bf16 [N,K]) ----
  __bf16* wsT  = wT;
  __bf16* wkT  = wsT + 16384;
  __bf16* wfT  = wkT + 4 * 16384;
  __bf16* wf1T = wfT + 16384;       // W_ff1 [128,512] -> [512,128]
  __bf16* wf2T = wf1T + 65536;      // W_ff2 [512,128] -> [128,512]
  k_transpose_cvt<<<(16384 + 255) / 256, 256, 0, st>>>(p.W_self, wsT, 128, 128);
  for (int i = 0; i < 4; ++i)
    k_transpose_cvt<<<(16384 + 255) / 256, 256, 0, st>>>(p.W_khops + (size_t)i * 16384,
                                                         wkT + (size_t)i * 16384, 128, 128);
  k_transpose_cvt<<<(16384 + 255) / 256, 256, 0, st>>>(p.W_fuse, wfT, 128, 128);
  k_transpose_cvt<<<(65536 + 255) / 256, 256, 0, st>>>(p.W_ff1, wf1T, 128, 512);
  k_transpose_cvt<<<(65536 + 255) / 256, 256, 0, st>>>(p.W_ff2, wf2T, 512, 128);

  const long nD = (long)n * 128;
  const long eT = (long)nE * 32;

  // prev_x = x @ W_self + b_self   (into pre)
  k_cvt_bf16<<<(nD + 255) / 256, 256, 0, st>>>(x, bfA, nD);
  launch_gemm(st, bfA, wsT, p.b_self, pre, nullptr, n, 128, 128, 0, 0);

  // k-hop scatter + per-hop linear, accumulated into pre
  const float* zcur = x;
  float* zbufs[2] = {z0, z1};
  for (int i = 0; i < 4; ++i) {
    float* zn = zbufs[i & 1];
    k_zero_f32<<<(nD + 255) / 256, 256, 0, st>>>(zn, nD);
    k_scatter_add<<<(eT + 255) / 256, 256, 0, st>>>(zcur, esrc, edst, zn, nE);
    k_cvt_bf16<<<(nD + 255) / 256, 256, 0, st>>>(zn, bfA, nD);
    launch_gemm(st, bfA, wkT + (size_t)i * 16384, p.b_khops + (size_t)i * 128,
                pre, nullptr, n, 128, 128, 1, 0);
    zcur = zn;
  }

  // edge_x = inc @ W_fuse + b_fuse, accumulated into pre
  k_cvt_bf16<<<(nD + 255) / 256, 256, 0, st>>>(inc, bfA, nD);
  launch_gemm(st, bfA, wfT, p.b_fuse, pre, nullptr, n, 128, 128, 1, 0);

  // h1 = LN(x + gelu(pre)); bf16 copy into bfA for FF1
  k_gelu_res_ln<<<n, 128, 0, st>>>(x, pre, p.ln1_g, p.ln1_b, h1, bfA);

  // H = gelu(h1 @ W_ff1 + b_ff1) -> bf16 [n,512]
  launch_gemm(st, bfA, wf1T, p.b_ff1, nullptr, bfH, n, 512, 128, 0, 1);

  // ff = H @ W_ff2 + b_ff2 -> fp32 (reuse z0; last k-hop result lives in z1)
  launch_gemm(st, bfH, wf2T, p.b_ff2, z0, nullptr, n, 128, 512, 0, 0);

  // out = LN(h1 + ff)
  k_res_ln<<<n, 128, 0, st>>>(h1, z0, p.ln2_g, p.ln2_b, out);
}

extern "C" void kernel_launch(void* const* d_in, const int* in_sizes, int n_in,
                              void* d_out, int out_size, void* d_ws, size_t ws_size,
                              hipStream_t stream) {
  const float* x   = (const float*)d_in[0];
  const float* lgx = (const float*)d_in[1];
  CoreParams np = unpack_params(d_in, 2);
  CoreParams ep = unpack_params(d_in, 16);
  const int* src  = (const int*)d_in[30];
  const int* dst  = (const int*)d_in[31];
  const int* lsrc = (const int*)d_in[32];
  const int* ldst = (const int*)d_in[33];

  const int nNodes = in_sizes[0] / 128;   // 40000
  const int nEdges = in_sizes[1] / 128;   // 400000
  const int nLg    = in_sizes[32];        // 800000
  const int NMAX   = nEdges;              // edge core is the big one

  // ---- carve workspace (~1.54 GB) ----
  char* w = (char*)d_ws;
  auto carve = [&](size_t bytes) -> void* {
    void* r = (void*)w;
    w += (bytes + 255) & ~(size_t)255;
    return r;
  };
  __bf16* bfA = (__bf16*)carve((size_t)NMAX * 128 * 2);   // bf16 A staging (D=128)
  __bf16* bfH = (__bf16*)carve((size_t)NMAX * 512 * 2);   // bf16 FF hidden [n,512]
  float*  z0  = (float*) carve((size_t)NMAX * 128 * 4);
  float*  z1  = (float*) carve((size_t)NMAX * 128 * 4);
  float*  pre = (float*) carve((size_t)NMAX * 128 * 4);
  float*  h1  = (float*) carve((size_t)NMAX * 128 * 4);
  float*  inc = (float*) carve((size_t)NMAX * 128 * 4);
  __bf16* wT  = (__bf16*)carve((size_t)229376 * 2);       // packed weights per core

  float* out_nodes = (float*)d_out;
  float* out_edges = out_nodes + (size_t)nNodes * 128;

  // ---- node core: inc = scatter lg_x to both endpoints ----
  {
    long nD = (long)nNodes * 128;
    long eT = (long)nEdges * 32;
    k_zero_f32<<<(unsigned)((nD + 255) / 256), 256, 0, stream>>>(inc, nD);
    k_inc_scatter<<<(unsigned)((eT + 255) / 256), 256, 0, stream>>>(lgx, src, dst, inc, nEdges);
  }
  run_core(stream, np, x, src, dst, nEdges, nNodes, inc,
           bfA, bfH, z0, z1, pre, h1, wT, out_nodes);

  // ---- edge core: inc = x[src] + x[dst] (gather) ----
  {
    long eT = (long)nEdges * 32;
    k_edge_inc<<<(unsigned)((eT + 255) / 256), 256, 0, stream>>>(x, src, dst, inc, nEdges);
  }
  run_core(stream, ep, lgx, lsrc, ldst, nLg, nEdges, inc,
           bfA, bfH, z0, z1, pre, h1, wT, out_edges);
}
